// SelectiveStateSpace_37950331027866
// MI455X (gfx1250) — matrix operand order repaired and checked by reading, not compiled, not run on hardware
//
#include <hip/hip_runtime.h>
#include <hip/hip_bf16.h>

// ---------------- problem constants ----------------
#define D_MODEL   1024
#define D_STATE   16
#define D_CONV    4
#define D_INNER   2048
#define DT_RANK   64
#define BATCH     2
#define SEQ_LEN   4096
#define ROWS      (BATCH * SEQ_LEN)        // 8192
#define XPROJ_N   (DT_RANK + 2 * D_STATE)  // 96

typedef __attribute__((ext_vector_type(16))) __bf16        v16bf;
typedef __attribute__((ext_vector_type(8)))  float         v8f;
typedef __attribute__((ext_vector_type(8)))  unsigned int  v8u;

// ---------------- helpers ----------------
__device__ __forceinline__ unsigned short f2bf(float f) {
    unsigned int u = __builtin_bit_cast(unsigned int, f);
    u += 0x7FFFu + ((u >> 16) & 1u);       // round-to-nearest-even
    return (unsigned short)(u >> 16);
}
__device__ __forceinline__ float bf2f(unsigned short s) {
    unsigned int u = ((unsigned int)s) << 16;
    return __builtin_bit_cast(float, u);
}
__device__ __forceinline__ float sigmoidf_fast(float x) {
    return 1.0f / (1.0f + __expf(-x));
}

// ---------------- f32 -> bf16 conversion ----------------
__global__ void k_f32_to_bf16(const float* __restrict__ src,
                              unsigned short* __restrict__ dst, int n) {
    for (int i = blockIdx.x * blockDim.x + threadIdx.x; i < n;
         i += gridDim.x * blockDim.x)
        dst[i] = f2bf(src[i]);
}

// ---------------- generic bf16 WMMA GEMM ----------------
// C[M,N] f32 = A[M,K](bf16 rm) * B[K,N](bf16 rm)
// Block tile 128x128, BK=32, 8 waves as 4(row)x2(col); wave = 32x64 out
//   = 2 A-frags x 4 B-frags = 8 WMMA / K-step.
// LDS: A row-major (pad 40 shorts = 80B, 16B-aligned rows),
//      B TRANSPOSED [col][k] so each lane's 16 K-shorts are contiguous
//      -> every fragment is two ds_load_b128.
// mode 0: C0[row*ldc+col] = v
// mode 1: col<nsplit -> C0 f32 ; else C1 bf16 (both ld nsplit)  [in_proj xc|z]
// mode 2: C0 = softplus(v + bias[col])                          [dt_proj]
__global__ __launch_bounds__(256)
void k_gemm_bf16(const unsigned short* __restrict__ A,
                 const unsigned short* __restrict__ B,
                 float* __restrict__ C0, unsigned short* __restrict__ C1,
                 const float* __restrict__ bias,
                 int M, int N, int K, int lda, int ldb, int ldc,
                 int nsplit, int mode) {
    __shared__ unsigned short a_sh[128 * 40];   // 128 rows x 32 K, stride 40
    __shared__ unsigned short b_shT[128 * 40];  // 128 cols x 32 K, stride 40

    const int tid  = threadIdx.x;
    const int lane = tid & 31;
    const int wid  = tid >> 5;                  // 0..7
    const int wr   = wid >> 1;                  // 0..3 row wave
    const int wc   = wid & 1;                   // 0..1 col wave
    const int rowBase = blockIdx.y * 128;
    const int colBase = blockIdx.x * 128;
    const int l15  = lane & 15;

    v8f c[2][4] = {};

    for (int k0 = 0; k0 < K; k0 += 32) {
        // ---- stage A tile (128x32 shorts) ----
        if (rowBase + 128 <= M && k0 + 32 <= K) {
            for (int i = tid; i < 512; i += 256) {           // 512 x uint4 (8 shorts)
                int r = i >> 2, seg = (i & 3) << 3;
                uint4 q = *(const uint4*)&A[(size_t)(rowBase + r) * lda + k0 + seg];
                *(uint4*)&a_sh[r * 40 + seg] = q;
            }
        } else {
            for (int i = tid; i < 128 * 32; i += 256) {
                int r = i >> 5, cc = i & 31;
                int gr = rowBase + r, gc = k0 + cc;
                a_sh[r * 40 + cc] = (gr < M && gc < K)
                    ? A[(size_t)gr * lda + gc] : (unsigned short)0;
            }
        }
        // ---- stage B tile transposed (32x128 -> [col][k]) ----
        if (colBase + 128 <= N && k0 + 32 <= K) {
            for (int i = tid; i < 2048; i += 256) {          // uint = 2 cols
                int r  = i >> 6;                              // k row 0..31
                int cp = (i & 63) << 1;                       // col pair
                unsigned int w = *(const unsigned int*)
                    &B[(size_t)(k0 + r) * ldb + colBase + cp];
                b_shT[cp * 40 + r]       = (unsigned short)(w & 0xFFFFu);
                b_shT[(cp + 1) * 40 + r] = (unsigned short)(w >> 16);
            }
        } else {
            for (int i = tid; i < 32 * 128; i += 256) {
                int r = i >> 7, cc = i & 127;
                int gr = k0 + r, gc = colBase + cc;
                b_shT[cc * 40 + r] = (gr < K && gc < N)
                    ? B[(size_t)gr * ldb + gc] : (unsigned short)0;
            }
        }
        if (k0 + 32 < K) {
            int pr = rowBase + (tid >> 1);
            if (pr < M) __builtin_prefetch(&A[(size_t)pr * lda + k0 + 32], 0, 1);
        }
        __syncthreads();

        // ---- A fragments: ISA 16-bit A 16x32 layout ----
        // lane<16: K = {0..7}+khalf (u0..u3), {16..23}+khalf (u4..u7); khalf=8 for lane>=16
        const int khalfA = (lane >> 4) << 3;
        v16bf af[2];
        #pragma unroll
        for (int mr = 0; mr < 2; ++mr) {
            int rA = (wr << 5) + (mr << 4) + l15;
            uint4 q0 = *(const uint4*)&a_sh[rA * 40 + khalfA];
            uint4 q1 = *(const uint4*)&a_sh[rA * 40 + khalfA + 16];
            v8u au;
            au[0]=q0.x; au[1]=q0.y; au[2]=q0.z; au[3]=q0.w;
            au[4]=q1.x; au[5]=q1.y; au[6]=q1.z; au[7]=q1.w;
            af[mr] = __builtin_bit_cast(v16bf, au);
        }
        // ---- B fragments + WMMA ----
        // lane<16: col=lane, K={0..7},{16..23} ; lane>=16: col=lane-16, K={8..15},{24..31}
        const int koffB = (lane >> 4) << 3;
        #pragma unroll
        for (int ns = 0; ns < 4; ++ns) {
            int colL = (wc << 6) + (ns << 4) + l15;
            uint4 b0 = *(const uint4*)&b_shT[colL * 40 + koffB];
            uint4 b1 = *(const uint4*)&b_shT[colL * 40 + koffB + 16];
            v8u bu;
            bu[0]=b0.x; bu[1]=b0.y; bu[2]=b0.z; bu[3]=b0.w;
            bu[4]=b1.x; bu[5]=b1.y; bu[6]=b1.z; bu[7]=b1.w;
            v16bf bf = __builtin_bit_cast(v16bf, bu);
            #pragma unroll
            for (int mr = 0; mr < 2; ++mr)
                c[mr][ns] = __builtin_amdgcn_wmma_f32_16x16x32_bf16(
                                false, af[mr], false, bf, (short)0,
                                c[mr][ns], false, false);
        }
        __syncthreads();
    }

    // ---- store: C/D layout lanes 0-15 (N=lane, M=r), lanes 16-31 (N=lane-16, M=r+8)
    const int rowHalf = (lane >> 4) << 3;
    #pragma unroll
    for (int mr = 0; mr < 2; ++mr) {
        #pragma unroll
        for (int ns = 0; ns < 4; ++ns) {
            int col = colBase + (wc << 6) + (ns << 4) + l15;
            #pragma unroll
            for (int r = 0; r < 8; ++r) {
                int row = rowBase + (wr << 5) + (mr << 4) + rowHalf + r;
                if (row < M && col < N) {
                    float v = c[mr][ns][r];
                    if (mode == 0) {
                        C0[(size_t)row * ldc + col] = v;
                    } else if (mode == 1) {
                        if (col < nsplit)
                            C0[(size_t)row * nsplit + col] = v;
                        else
                            C1[(size_t)row * nsplit + (col - nsplit)] = f2bf(v);
                    } else {
                        v += bias[col];
                        v = (v > 20.0f) ? v : log1pf(__expf(v));
                        C0[(size_t)row * ldc + col] = v;
                    }
                }
            }
        }
    }
}

// ---------------- depthwise causal conv (k=4) + SiLU -> bf16 ----------------
__global__ void k_conv_silu(const float* __restrict__ xc,
                            const float* __restrict__ conv_w,
                            const float* __restrict__ conv_b,
                            unsigned short* __restrict__ xsbf) {
    int i = blockIdx.x * blockDim.x + threadIdx.x;
    if (i >= ROWS * D_INNER) return;
    int col = i % D_INNER;
    int row = i / D_INNER;
    int t = row % SEQ_LEN;
    int b = row / SEQ_LEN;
    float acc = conv_b[col];
    #pragma unroll
    for (int k = 0; k < D_CONV; ++k) {
        int tt = t + k - (D_CONV - 1);
        if (tt >= 0)
            acc += xc[((size_t)(b * SEQ_LEN + tt)) * D_INNER + col] * conv_w[col * D_CONV + k];
    }
    float s = acc * sigmoidf_fast(acc);
    xsbf[i] = f2bf(s);
}

// ---------------- selective scan: one thread per (b,d) channel ----------------
__global__ __launch_bounds__(256)
void k_scan(const float* __restrict__ dt,
            const unsigned short* __restrict__ xsbf,
            const float* __restrict__ xdbl,       // [ROWS,96]; B at 64..79, C at 80..95
            const float* __restrict__ A_log,
            float* __restrict__ y) {
    int ch = blockIdx.x * blockDim.x + threadIdx.x;
    if (ch >= BATCH * D_INNER) return;
    int b = ch / D_INNER, d = ch % D_INNER;

    float Ad[D_STATE], h[D_STATE];
    #pragma unroll
    for (int n = 0; n < D_STATE; ++n) {
        Ad[n] = -__expf(A_log[d * D_STATE + n]);
        h[n] = 0.0f;
    }

    for (int t = 0; t < SEQ_LEN; ++t) {
        size_t row = (size_t)b * SEQ_LEN + t;
        float dtv = dt[row * D_INNER + d];
        float u   = bf2f(xsbf[row * D_INNER + d]);
        const float4* Bp = (const float4*)(xdbl + row * XPROJ_N + DT_RANK);
        const float4* Cp = (const float4*)(xdbl + row * XPROJ_N + DT_RANK + D_STATE);
        float Bv[D_STATE], Cv[D_STATE];
        #pragma unroll
        for (int q = 0; q < 4; ++q) {
            float4 bb = Bp[q], cc = Cp[q];
            Bv[q*4+0]=bb.x; Bv[q*4+1]=bb.y; Bv[q*4+2]=bb.z; Bv[q*4+3]=bb.w;
            Cv[q*4+0]=cc.x; Cv[q*4+1]=cc.y; Cv[q*4+2]=cc.z; Cv[q*4+3]=cc.w;
        }
        float du = dtv * u;
        float acc = 0.0f;
        #pragma unroll
        for (int n = 0; n < D_STATE; ++n) {
            float dA = __expf(dtv * Ad[n]);
            h[n] = dA * h[n] + du * Bv[n];
            acc += h[n] * Cv[n];
        }
        y[row * D_INNER + d] = acc;
    }
}

// ---------------- gating: ybf = (y + D*xs) * silu(z) ----------------
__global__ void k_gate(const float* __restrict__ y,
                       const unsigned short* __restrict__ xsbf,
                       const unsigned short* __restrict__ zbf,
                       const float* __restrict__ Dp,
                       unsigned short* __restrict__ ybf) {
    int i = blockIdx.x * blockDim.x + threadIdx.x;
    if (i >= ROWS * D_INNER) return;
    int col = i % D_INNER;
    float v = y[i] + Dp[col] * bf2f(xsbf[i]);
    float z = bf2f(zbf[i]);
    v *= z * sigmoidf_fast(z);
    ybf[i] = f2bf(v);
}

// ---------------- host side ----------------
extern "C" void kernel_launch(void* const* d_in, const int* in_sizes, int n_in,
                              void* d_out, int out_size, void* d_ws, size_t ws_size,
                              hipStream_t stream) {
    const float* x         = (const float*)d_in[0];  // [2,4096,1024]
    const float* in_proj_w = (const float*)d_in[1];  // [1024,4096]
    const float* conv_w    = (const float*)d_in[2];  // [2048,4]
    const float* conv_b    = (const float*)d_in[3];  // [2048]
    const float* x_proj_w  = (const float*)d_in[4];  // [2048,96]
    const float* dt_proj_w = (const float*)d_in[5];  // [64,2048]
    const float* dt_proj_b = (const float*)d_in[6];  // [2048]
    const float* A_log     = (const float*)d_in[7];  // [2048,16]
    const float* D_param   = (const float*)d_in[8];  // [2048]
    const float* out_proj_w= (const float*)d_in[9];  // [2048,1024]
    float* out = (float*)d_out;                      // [2,4096,1024]

    char* ws = (char*)d_ws;
    size_t off = 0;
    auto alloc = [&](size_t bytes) -> char* {
        char* p = ws + off;
        off += (bytes + 255) & ~(size_t)255;
        return p;
    };
    unsigned short* xbf     = (unsigned short*)alloc((size_t)ROWS * D_MODEL * 2);
    unsigned short* winbf   = (unsigned short*)alloc((size_t)D_MODEL * 2 * D_INNER * 2);
    unsigned short* xprojbf = (unsigned short*)alloc((size_t)D_INNER * XPROJ_N * 2);
    unsigned short* dtwbf   = (unsigned short*)alloc((size_t)DT_RANK * D_INNER * 2);
    unsigned short* outwbf  = (unsigned short*)alloc((size_t)D_INNER * D_MODEL * 2);
    float*          xc      = (float*)alloc((size_t)ROWS * D_INNER * 4);
    unsigned short* zbf     = (unsigned short*)alloc((size_t)ROWS * D_INNER * 2);
    unsigned short* xsbf    = (unsigned short*)alloc((size_t)ROWS * D_INNER * 2);
    float*          xdbl    = (float*)alloc((size_t)ROWS * XPROJ_N * 4);
    unsigned short* xdblbf  = (unsigned short*)alloc((size_t)ROWS * XPROJ_N * 2);
    float*          dt      = (float*)alloc((size_t)ROWS * D_INNER * 4);
    float*          yacc    = (float*)alloc((size_t)ROWS * D_INNER * 4);
    unsigned short* ybf     = (unsigned short*)alloc((size_t)ROWS * D_INNER * 2);

    auto cvt = [&](const float* s, unsigned short* d, int n) {
        int g = (n + 255) / 256; if (g > 4096) g = 4096;
        hipLaunchKernelGGL(k_f32_to_bf16, dim3(g), dim3(256), 0, stream, s, d, n);
    };

    // 1. downconvert activations + weights
    cvt(x,          xbf,     ROWS * D_MODEL);
    cvt(in_proj_w,  winbf,   D_MODEL * 2 * D_INNER);
    cvt(x_proj_w,   xprojbf, D_INNER * XPROJ_N);
    cvt(dt_proj_w,  dtwbf,   DT_RANK * D_INNER);
    cvt(out_proj_w, outwbf,  D_INNER * D_MODEL);

    // 2. in_proj GEMM, split epilogue -> xc (f32) | z (bf16)
    hipLaunchKernelGGL(k_gemm_bf16, dim3((2 * D_INNER + 127) / 128, ROWS / 128),
                       dim3(256), 0, stream,
                       xbf, winbf, xc, zbf, (const float*)nullptr,
                       ROWS, 2 * D_INNER, D_MODEL, D_MODEL, 2 * D_INNER, 0,
                       D_INNER, 1);

    // 3. depthwise conv + SiLU -> xsbf
    hipLaunchKernelGGL(k_conv_silu, dim3(ROWS * D_INNER / 256), dim3(256), 0, stream,
                       xc, conv_w, conv_b, xsbf);

    // 4. x_proj GEMM -> xdbl f32 [ROWS,96]
    hipLaunchKernelGGL(k_gemm_bf16, dim3(1, ROWS / 128),
                       dim3(256), 0, stream,
                       xsbf, xprojbf, xdbl, (unsigned short*)nullptr, (const float*)nullptr,
                       ROWS, XPROJ_N, D_INNER, D_INNER, XPROJ_N, XPROJ_N, 0, 0);

    // 5. convert x_dbl to bf16 (dt_low operand, lda=96)
    cvt(xdbl, xdblbf, ROWS * XPROJ_N);

    // 6. dt_proj GEMM + bias + softplus -> dt f32
    hipLaunchKernelGGL(k_gemm_bf16, dim3(D_INNER / 128, ROWS / 128),
                       dim3(256), 0, stream,
                       xdblbf, dtwbf, dt, (unsigned short*)nullptr, dt_proj_b,
                       ROWS, D_INNER, DT_RANK, XPROJ_N, D_INNER, D_INNER, 0, 2);

    // 7. selective scan -> yacc
    hipLaunchKernelGGL(k_scan, dim3((BATCH * D_INNER) / 256), dim3(256), 0, stream,
                       dt, xsbf, xdbl, A_log, yacc);

    // 8. gating -> ybf
    hipLaunchKernelGGL(k_gate, dim3(ROWS * D_INNER / 256), dim3(256), 0, stream,
                       yacc, xsbf, zbf, D_param, ybf);

    // 9. out_proj GEMM -> d_out f32
    hipLaunchKernelGGL(k_gemm_bf16, dim3(D_MODEL / 128, ROWS / 128),
                       dim3(256), 0, stream,
                       ybf, outwbf, out, (unsigned short*)nullptr, (const float*)nullptr,
                       ROWS, D_MODEL, D_INNER, D_INNER, D_MODEL, D_MODEL, 0, 0);
}
